// LogicNetwork_84121229460090
// MI455X (gfx1250) — matrix operand order repaired and checked by reading, not compiled, not run on hardware
//
#include <hip/hip_runtime.h>
#include <hip/hip_bf16.h>
#include <math.h>

// ---------------------------------------------------------------------------
// LogicNetwork on MI455X (gfx1250): all GEMMs via v_wmma_f32_16x16x32_f16.
// One wave (32 lanes) owns a 16-row batch tile; 8 waves / block -> 128 rows.
// Weights staged once per block into LDS as f16 column-major so B fragments
// are contiguous ds_load_b128; activations ping-pong through per-wave f16
// LDS tiles (same-wave LDS ops are in-order on CDNA5, no barriers needed).
// ~130 KB LDS/block -> 2 workgroups/WGP for cross-wave gather latency hiding.
// Late embedding gathers are warmed with global_prefetch_b8 at kernel start.
// ---------------------------------------------------------------------------

typedef _Float16 v8h  __attribute__((ext_vector_type(8)));
typedef _Float16 v16h __attribute__((ext_vector_type(16)));
typedef _Float16 h2   __attribute__((ext_vector_type(2)));
typedef float    v8f  __attribute__((ext_vector_type(8)));

#define EMBED_DIM 64
#define BATCH     131072
#define ROWS_PER_WAVE 16
#define WAVES_PER_BLOCK 8
#define ROWS_PER_BLOCK (ROWS_PER_WAVE * WAVES_PER_BLOCK)
#define NTHREADS 256

// LDS half-precision weight offsets (column-major: W[n*K + k])
#define OFF_NW1 0          // 64x64
#define OFF_NW2 4096       // 64x64
#define OFF_AW1 8192       // 128x64
#define OFF_AW2 16384      // 64x64
#define OFF_OW1 20480      // 128x64
#define OFF_OW2 28672      // 64x64
#define W_HALFS 32768

#define TILE_HALFS (16 * 64)            // one 16x64 f16 activation tile
#define WAVE_HALFS (4 * TILE_HALFS)     // TA, TB, TC, H per wave
#define SMEM_BYTES ((64 + 6 * 64) * 4 + W_HALFS * 2 + WAVES_PER_BLOCK * WAVE_HALFS * 2)

// Build a v16h fragment from two aligned 8-half LDS chunks (2x ds_load_b128).
__device__ __forceinline__ v16h frag16(const _Float16* p0, const _Float16* p1) {
  v8h lo = *(const v8h*)p0;
  v8h hi = *(const v8h*)p1;
  return __builtin_shufflevector(lo, hi, 0, 1, 2, 3, 4, 5, 6, 7,
                                         8, 9, 10, 11, 12, 13, 14, 15);
}

// A fragment (16x32 f16, ISA layout): lanes 0-15 hold K={kl..kl+7, kl+16..+23},
// lanes 16-31 hold the +8 chunks. X is a row-major 16x64 f16 tile.
__device__ __forceinline__ v16h load_a(const _Float16* X, int m, int kl, int hs) {
  const _Float16* base = X + m * 64 + kl + hs * 8;
  return frag16(base, base + 16);
}

// B fragment (32x16 f16): lane holds column n, K = {kb+8*hs..+7, kb+16+8*hs..+7},
// matching the A layout; W is stored column-major with stride Kdim.
__device__ __forceinline__ v16h load_b(const _Float16* W, int Kdim, int n, int kb, int hs) {
  const _Float16* base = W + n * Kdim + kb + hs * 8;
  return frag16(base, base + 16);
}

// One 16xKIN @ KINx64 GEMM for a wave. X0 = cols 0..63, X1 = cols 64..127
// (KIN==128 => concatenated input, each 32-wide K step stays in one tile).
// Bias always added; optional leaky-ReLU (fmax form: slope 0.01 > 0 so
// leaky(x) == max(x, 0.01x), avoiding cmp/cndmask VCC chains).
template <int KIN, bool LEAKY, bool STORE>
__device__ __forceinline__ void gemm(const _Float16* X0, const _Float16* X1,
                                     const _Float16* W, const float* bias,
                                     _Float16* Y, v8f* accOut, int lane) {
  const int l15 = lane & 15;
  const int hs = lane >> 4;
#pragma unroll
  for (int nt = 0; nt < 4; ++nt) {
    v8f c = {0.f, 0.f, 0.f, 0.f, 0.f, 0.f, 0.f, 0.f};
#pragma unroll
    for (int kb = 0; kb < KIN; kb += 32) {
      const _Float16* Xt = (kb < 64) ? X0 : X1;
      v16h a = load_a(Xt, l15, kb & 63, hs);
      v16h b = load_b(W, KIN, nt * 16 + l15, kb, hs);
      c = __builtin_amdgcn_wmma_f32_16x16x32_f16(false, a, false, b,
                                                 (short)0, c, false, false);
    }
    const int n = nt * 16 + l15;
    const float bv = bias[n];
#pragma unroll
    for (int r = 0; r < 8; ++r) {
      float v = c[r] + bv;
      if (LEAKY) v = fmaxf(v, 0.01f * v);
      c[r] = v;
    }
    if (STORE) {
      const int mbase = hs * 8;
#pragma unroll
      for (int r = 0; r < 8; ++r)
        Y[(mbase + r) * 64 + n] = (_Float16)c[r];
    } else {
      accOut[nt] = c;
    }
  }
}

// Two-layer MLP: h = leaky(X@W1+b1); Y = h@W2+b2  (Y as f16 LDS tile).
template <int KIN>
__device__ __forceinline__ void mlp(const _Float16* X0, const _Float16* X1,
                                    const _Float16* W1, const float* b1,
                                    const _Float16* W2, const float* b2,
                                    _Float16* H, _Float16* Y, int lane) {
  gemm<KIN, true, true>(X0, X1, W1, b1, H, nullptr, lane);
  gemm<64, false, true>(H, nullptr, W2, b2, Y, nullptr, lane);
}

// Final MLP variant: keep f32 accumulators in registers for the cosine head.
__device__ __forceinline__ void mlp_final(const _Float16* X0, const _Float16* X1,
                                          const _Float16* W1, const float* b1,
                                          const _Float16* W2, const float* b2,
                                          _Float16* H, v8f acc[4], int lane) {
  gemm<128, true, true>(X0, X1, W1, b1, H, nullptr, lane);
  gemm<64, false, false>(H, nullptr, W2, b2, nullptr, acc, lane);
}

// Warm one 16-row gather: lanes 0-15 pick rows, lanes 16-31 prefetch the
// second 128B half of the 256B embedding row. Emits global_prefetch_b8,
// no LOADcnt / register cost.
__device__ __forceinline__ void prefetch_rows(const float* emb, const int* idx,
                                              int stride, int rowBase, int lane) {
  const int r = lane & 15;
  const int half = (lane >> 4) * 128;
  const int id = idx[(size_t)(rowBase + r) * stride];
  const char* p = (const char*)(emb + (size_t)id * EMBED_DIM) + half;
  __builtin_prefetch(p, 0, 3);
}

// Gather 16 embedding rows into a 16x64 f16 tile. Each lane loads float2
// (8B) => a full 256B row per 32-lane wave, fully coalesced.
__device__ __forceinline__ void gather16(_Float16* T, const float* emb,
                                         const int* idx, int stride,
                                         int rowBase, int lane) {
#pragma unroll 4
  for (int r = 0; r < 16; ++r) {
    const int id = idx[(size_t)(rowBase + r) * stride];
    const float2* src = (const float2*)(emb + (size_t)id * EMBED_DIM);
    const float2 v = src[lane];
    h2 w;
    w.x = (_Float16)v.x;
    w.y = (_Float16)v.y;
    *(h2*)(T + r * 64 + lane * 2) = w;
  }
}

// Cosine head: acc holds encoded 16x64 tile (lane = col n, 8 rows). Reduce
// dot/norm across the 16-lane half with xor butterflies, write 16 preds.
__device__ __forceinline__ void predict_store(const v8f acc[4], const float* s_tv,
                                              float tvnorm, float* out,
                                              int rowBase, int lane) {
  const int l15 = lane & 15;
  const int hs = lane >> 4;
#pragma unroll
  for (int r = 0; r < 8; ++r) {
    float d = 0.f, q = 0.f;
#pragma unroll
    for (int nt = 0; nt < 4; ++nt) {
      const float v = acc[nt][r];
      d += v * s_tv[nt * 16 + l15];
      q += v * v;
    }
#pragma unroll
    for (int m = 8; m >= 1; m >>= 1) {  // masks < 16: stays within each half
      d += __shfl_xor(d, m, 32);
      q += __shfl_xor(q, m, 32);
    }
    if (l15 == r) {
      const float denom = fmaxf(sqrtf(q), 1e-8f) * fmaxf(tvnorm, 1e-8f);
      out[rowBase + hs * 8 + r] = d / denom * 10.0f;
    }
  }
}

// Stage a KDIMx64 f32 row-major weight into LDS f16 column-major.
__device__ __forceinline__ void stage_w(const float* G, _Float16* L, int Kdim,
                                        int tid) {
  const int total = Kdim * 64;
  for (int i = tid; i < total; i += NTHREADS) {
    const int k = i >> 6, n = i & 63;
    L[n * Kdim + k] = (_Float16)G[i];
  }
}

__global__ __launch_bounds__(NTHREADS, 1) void logicnet_kernel(
    const int* __restrict__ seq, const int* __restrict__ pos_t,
    const int* __restrict__ neg_t, const float* __restrict__ emb,
    const float* __restrict__ true_vec,
    const float* __restrict__ nW1, const float* __restrict__ nb1,
    const float* __restrict__ nW2, const float* __restrict__ nb2,
    const float* __restrict__ aW1, const float* __restrict__ ab1,
    const float* __restrict__ aW2, const float* __restrict__ ab2,
    const float* __restrict__ oW1, const float* __restrict__ ob1,
    const float* __restrict__ oW2, const float* __restrict__ ob2,
    float* __restrict__ out) {
  extern __shared__ __align__(16) char smem[];
  float* s_tv = (float*)smem;                  // 64 f32
  float* s_b = s_tv + 64;                      // 6 x 64 f32 biases
  _Float16* s_w = (_Float16*)(s_b + 6 * 64);   // 32768 f16 weights
  _Float16* tiles = s_w + W_HALFS;             // per-wave activation tiles

  const int tid = threadIdx.x;
  const int lane = tid & 31;
  const int wave = tid >> 5;
  const int rowBase = blockIdx.x * ROWS_PER_BLOCK + wave * ROWS_PER_WAVE;

  // ---- warm late-used embedding rows in L2/WGP$ (global_prefetch_b8) ----
  prefetch_rows(emb, seq + 2, 5, rowBase, lane);
  prefetch_rows(emb, seq + 3, 5, rowBase, lane);
  prefetch_rows(emb, seq + 4, 5, rowBase, lane);
  prefetch_rows(emb, pos_t, 1, rowBase, lane);
  prefetch_rows(emb, neg_t, 1, rowBase, lane);

  // ---- stage constants (block-cooperative), one barrier total ----
  if (tid < 64) s_tv[tid] = true_vec[tid];
  if (tid >= 64 && tid < 128) {
    const int i = tid - 64;
    s_b[i] = nb1[i];        s_b[64 + i] = nb2[i];
    s_b[128 + i] = ab1[i];  s_b[192 + i] = ab2[i];
    s_b[256 + i] = ob1[i];  s_b[320 + i] = ob2[i];
  }
  stage_w(nW1, s_w + OFF_NW1, 64, tid);
  stage_w(nW2, s_w + OFF_NW2, 64, tid);
  stage_w(aW1, s_w + OFF_AW1, 128, tid);
  stage_w(aW2, s_w + OFF_AW2, 64, tid);
  stage_w(oW1, s_w + OFF_OW1, 128, tid);
  stage_w(oW2, s_w + OFF_OW2, 64, tid);
  __syncthreads();

  float tq = 0.f;
#pragma unroll
  for (int i = 0; i < 64; ++i) tq += s_tv[i] * s_tv[i];
  const float tvnorm = sqrtf(tq);

  _Float16* TA = tiles + wave * WAVE_HALFS;
  _Float16* TB = TA + TILE_HALFS;
  _Float16* TC = TB + TILE_HALFS;
  _Float16* H  = TC + TILE_HALFS;

  const _Float16 *WN1 = s_w + OFF_NW1, *WN2 = s_w + OFF_NW2;
  const _Float16 *WA1 = s_w + OFF_AW1, *WA2 = s_w + OFF_AW2;
  const _Float16 *WO1 = s_w + OFF_OW1, *WO2 = s_w + OFF_OW2;
  const float *BN1 = s_b, *BN2 = s_b + 64, *BA1 = s_b + 128, *BA2 = s_b + 192;
  const float *BO1 = s_b + 256, *BO2 = s_b + 320;

  // ---- DAG (per-wave, in-order LDS keeps RAW safe without barriers) ----
  gather16(TA, emb, seq + 1, 5, rowBase, lane);                 // e1
  mlp<64>(TA, nullptr, WN1, BN1, WN2, BN2, H, TB, lane);        // TB = NOT(e1)
  gather16(TA, emb, seq + 0, 5, rowBase, lane);                 // e0
  mlp<128>(TA, TB, WA1, BA1, WA2, BA2, H, TC, lane);            // TC = L5 = AND(e0, NOT e1)
  gather16(TA, emb, seq + 2, 5, rowBase, lane);                 // e2
  gather16(TB, emb, seq + 3, 5, rowBase, lane);                 // e3
  mlp<128>(TA, TB, WO1, BO1, WO2, BO2, H, TA, lane);            // TA = L6 = OR(e2, e3)
  mlp<128>(TC, TA, WA1, BA1, WA2, BA2, H, TB, lane);            // TB = L7 = AND(L5, L6)
  mlp<64>(TB, nullptr, WN1, BN1, WN2, BN2, H, TC, lane);        // TC = NOT(L7)
  gather16(TA, emb, seq + 4, 5, rowBase, lane);                 // e4
  mlp<128>(TC, TA, WO1, BO1, WO2, BO2, H, TB, lane);            // TB = out = OR(NOT L7, e4)
  mlp<64>(TB, nullptr, WN1, BN1, WN2, BN2, H, TC, lane);        // TC = enc_not = NOT(out)

  v8f acc[4];
  gather16(TA, emb, pos_t, 1, rowBase, lane);                   // pos_e
  mlp_final(TC, TA, WO1, BO1, WO2, BO2, H, acc, lane);          // encoded_pos
  predict_store(acc, s_tv, tvnorm, out, rowBase, lane);

  gather16(TA, emb, neg_t, 1, rowBase, lane);                   // neg_e
  mlp_final(TC, TA, WO1, BO1, WO2, BO2, H, acc, lane);          // encoded_neg
  predict_store(acc, s_tv, tvnorm, out + BATCH, rowBase, lane);
}

extern "C" void kernel_launch(void* const* d_in, const int* in_sizes, int n_in,
                              void* d_out, int out_size, void* d_ws, size_t ws_size,
                              hipStream_t stream) {
  (void)in_sizes; (void)n_in; (void)out_size; (void)d_ws; (void)ws_size;
  const int* seq = (const int*)d_in[0];
  const int* pos_t = (const int*)d_in[1];
  const int* neg_t = (const int*)d_in[2];
  const float* emb = (const float*)d_in[3];
  const float* tv = (const float*)d_in[4];
  const float* nW1 = (const float*)d_in[5];
  const float* nb1 = (const float*)d_in[6];
  const float* nW2 = (const float*)d_in[7];
  const float* nb2 = (const float*)d_in[8];
  const float* aW1 = (const float*)d_in[9];
  const float* ab1 = (const float*)d_in[10];
  const float* aW2 = (const float*)d_in[11];
  const float* ab2 = (const float*)d_in[12];
  const float* oW1 = (const float*)d_in[13];
  const float* ob1 = (const float*)d_in[14];
  const float* oW2 = (const float*)d_in[15];
  const float* ob2 = (const float*)d_in[16];

  const int grid = BATCH / ROWS_PER_BLOCK;  // 1024
  logicnet_kernel<<<grid, NTHREADS, SMEM_BYTES, stream>>>(
      seq, pos_t, neg_t, emb, tv, nW1, nb1, nW2, nb2, aW1, ab1, aW2, ab2,
      oW1, ob1, oW2, ob2, (float*)d_out);
}
